// ContrastiveSWM_21406117003666
// MI455X (gfx1250) — compile-verified, matrix operand order checked by reading, not run on hardware
//
#include <hip/hip_runtime.h>
#include <hip/hip_bf16.h>

// ---------------------------------------------------------------------------
// ContrastiveSWM transition GNN for MI455X (gfx1250, wave32, WMMA f16->f32)
// GEMMs: LDS-staged weight panels (async-to-LDS when available), 32x64 tiles
// per wave, 8 v_wmma per 32-K chunk.
// ---------------------------------------------------------------------------

typedef __attribute__((ext_vector_type(16))) _Float16 v16h;
typedef __attribute__((ext_vector_type(8)))  _Float16 v8h;
typedef __attribute__((ext_vector_type(8)))  float    v8f;
typedef int v4i_vec __attribute__((vector_size(16)));

#define BB 1024
#define NN 8
#define DD 256
#define HH 512
#define AA 4
#define NE (BB * 42)          // 43008 edges
#define NR (BB * NN)          // 8192 node rows
#define KE (3 * DD)           // 768  edge-MLP input dim
#define KN_RAW (HH + DD + AA) // 772  node-MLP input dim
#define KN_PAD 800            // padded to multiple of 32

#define SLAB 256              // K-slab staged in LDS (halves)
#define LDL  (SLAB + 8)       // padded LDS row stride -> conflict-free b-frag reads

#if defined(__gfx1250__) && __has_builtin(__builtin_amdgcn_global_load_async_to_lds_b128)
#define USE_ASYNC_LDS 1
typedef __attribute__((address_space(1))) v4i_vec* gptr_b128;
typedef __attribute__((address_space(3))) v4i_vec* lptr_b128;
#else
#define USE_ASYNC_LDS 0
#endif

// -------------------- fragment helpers (16-bit A/B VGPR layout) ------------
// 16-bit 16x32 operand, lane L<16: row M=L, halves = K{k0..k0+7, k0+16..k0+23}
//                       lane L>=16: row M=L-16, halves = K{k0+8..15, k0+24..31}
__device__ __forceinline__ v16h frag_from(const _Float16* p) {
    v8h lo = *reinterpret_cast<const v8h*>(p);
    v8h hi = *reinterpret_cast<const v8h*>(p + 16);
    return __builtin_shufflevector(lo, hi, 0,1,2,3,4,5,6,7,8,9,10,11,12,13,14,15);
}
__device__ __forceinline__ v16h load_frag_g(const _Float16* base, int ld,
                                            int r0, int k0, int lane) {
    return frag_from(base + (size_t)(r0 + (lane & 15)) * ld + k0 + ((lane & 16) ? 8 : 0));
}
__device__ __forceinline__ v16h load_frag_l(const _Float16* base,
                                            int r0, int k0, int lane) {
    return frag_from(base + (r0 + (lane & 15)) * LDL + k0 + ((lane & 16) ? 8 : 0));
}

// -------------------- WMMA GEMM with LDS-staged weight panel ---------------
// Out[M,N] = act(A[M,K] @ W^T + bias), Wt[N,K] row-major f16.
// Block = 256 threads = 8 waves; block tile = 256M x 64N; wave tile = 32M x 64N.
// M % 256 == 0, N % 64 == 0, K % 32 == 0.
__global__ __launch_bounds__(256) void gemm_wmma_kernel(
        const _Float16* __restrict__ A, const _Float16* __restrict__ Wt,
        const float* __restrict__ bias, float* __restrict__ outF,
        _Float16* __restrict__ outH, int M, int N, int K, int relu) {
    __shared__ _Float16 bpanel[64 * LDL];   // 33,792 B

    const int tid  = threadIdx.x;
    const int lane = tid & 31;
    const int wave = tid >> 5;
    const int m0   = (blockIdx.y * 8 + wave) * 32;
    const int n0   = blockIdx.x * 64;

    v8f acc[2][4];
#pragma unroll
    for (int mt = 0; mt < 2; ++mt)
#pragma unroll
        for (int t = 0; t < 4; ++t)
#pragma unroll
            for (int v = 0; v < 8; ++v) acc[mt][t][v] = 0.0f;

    for (int ks = 0; ks < K; ks += SLAB) {
        const int kl  = (K - ks < SLAB) ? (K - ks) : SLAB;   // multiple of 32
        const int cpr = kl >> 3;                              // 16B chunks per row

        // ---- stage Wt[n0..n0+64) x [ks..ks+kl) into LDS (padded rows) ----
        for (int c = tid; c < 64 * cpr; c += 256) {
            const int rn = c / cpr;
            const int kc = c - rn * cpr;
            const _Float16* g = Wt + (size_t)(n0 + rn) * K + ks + kc * 8;
            _Float16* l = bpanel + rn * LDL + kc * 8;
#if USE_ASYNC_LDS
            __builtin_amdgcn_global_load_async_to_lds_b128(
                (gptr_b128)g, (lptr_b128)l, 0, 0);
#else
            *reinterpret_cast<v8h*>(l) = *reinterpret_cast<const v8h*>(g);
#endif
        }
#if USE_ASYNC_LDS
#if __has_builtin(__builtin_amdgcn_s_wait_asynccnt)
        __builtin_amdgcn_s_wait_asynccnt(0);
#else
        asm volatile("s_wait_asynccnt 0x0" ::: "memory");
#endif
#endif
        __syncthreads();

        // ---- compute over the slab: A from global, B from LDS ----
        for (int k0 = 0; k0 < kl; k0 += 32) {
            v16h a0 = load_frag_g(A, K, m0,      ks + k0, lane);
            v16h a1 = load_frag_g(A, K, m0 + 16, ks + k0, lane);
#pragma unroll
            for (int t = 0; t < 4; ++t) {
                v16h b = load_frag_l(bpanel, t * 16, k0, lane);
                acc[0][t] = __builtin_amdgcn_wmma_f32_16x16x32_f16(
                    false, a0, false, b, (short)0, acc[0][t], false, false);
                acc[1][t] = __builtin_amdgcn_wmma_f32_16x16x32_f16(
                    false, a1, false, b, (short)0, acc[1][t], false, false);
            }
        }
        __syncthreads();
    }

    // C/D layout: VGPR v, lane L: row = v + (L&16 ? 8 : 0), col = L&15
    const int rowoff = (lane & 16) ? 8 : 0;
    const int coloff = lane & 15;
#pragma unroll
    for (int mt = 0; mt < 2; ++mt) {
#pragma unroll
        for (int t = 0; t < 4; ++t) {
            const int col = n0 + t * 16 + coloff;
            const float bv = bias ? bias[col] : 0.0f;
#pragma unroll
            for (int v = 0; v < 8; ++v) {
                const int row = m0 + mt * 16 + v + rowoff;
                float val = acc[mt][t][v] + bv;
                if (relu) val = fmaxf(val, 0.0f);
                if (outF) outF[(size_t)row * N + col] = val;
                if (outH) outH[(size_t)row * N + col] = (_Float16)val;
            }
        }
    }
}

// -------------------- LayerNorm (biased var, eps=1e-5) + relu -> f16 -------
__global__ void ln_relu_kernel(const float* __restrict__ X,
                               const float* __restrict__ g,
                               const float* __restrict__ bn,
                               _Float16* __restrict__ Y) {
    const int row = blockIdx.x;
    const float* x = X + (size_t)row * HH;
    __shared__ float s1[256], s2[256];
    const int tid = threadIdx.x;
    const float a = x[tid];
    const float b = x[tid + 256];
    s1[tid] = a + b;
    s2[tid] = a * a + b * b;
    __syncthreads();
    for (int off = 128; off > 0; off >>= 1) {
        if (tid < off) { s1[tid] += s1[tid + off]; s2[tid] += s2[tid + off]; }
        __syncthreads();
    }
    const float mean = s1[0] * (1.0f / HH);
    const float var  = s2[0] * (1.0f / HH) - mean * mean;
    const float inv  = rsqrtf(var + 1e-5f);
    float y0 = fmaxf((a - mean) * inv * g[tid] + bn[tid], 0.0f);
    float y1 = fmaxf((b - mean) * inv * g[tid + 256] + bn[tid + 256], 0.0f);
    Y[(size_t)row * HH + tid]       = (_Float16)y0;
    Y[(size_t)row * HH + tid + 256] = (_Float16)y1;
}

// -------------------- weight transpose + cast: W[K,N] f32 -> Wt[N,Kpad] f16
__global__ void wtrans_kernel(const float* __restrict__ W, _Float16* __restrict__ Wt,
                              int K, int N, int Kpad) {
    const int idx = blockIdx.x * blockDim.x + threadIdx.x;
    if (idx >= N * Kpad) return;
    const int n = idx / Kpad, kp = idx % Kpad;
    Wt[idx] = (kp < K) ? (_Float16)W[(size_t)kp * N + n] : (_Float16)0.0f;
}

// -------------------- edge-MLP input assembly ------------------------------
// Faithful edge list: n = N-1 = 7 nodes, offset b*7 (reference quirk).
__global__ void edge_input_kernel(const float* __restrict__ flat,
                                  const float* __restrict__ node,
                                  _Float16* __restrict__ Xe, int round0) {
    const size_t idx = (size_t)blockIdx.x * blockDim.x + threadIdx.x;
    if (idx >= (size_t)NE * KE) return;
    const int e = (int)(idx / KE);
    const int c = (int)(idx % KE);
    const int b = e / 42, p = e % 42;
    const int i = p / 6;
    const int j6 = p % 6;
    const int j = j6 + (j6 >= i ? 1 : 0);
    const int row = b * 7 + i;
    const int col = b * 7 + j;
    float v;
    if (c < DD)            v = flat[(size_t)row * DD + c];
    else if (c < 2 * DD)   v = round0 ? 0.0f : node[(size_t)row * DD + (c - DD)];
    else                   v = flat[(size_t)col * DD + (c - 2 * DD)];
    Xe[idx] = (_Float16)v;
}

// -------------------- segment sum (contiguous: 6 edges per node slot) ------
__global__ void agg_kernel(const float* __restrict__ ef, float* __restrict__ agg) {
    const int idx = blockIdx.x * blockDim.x + threadIdx.x;
    if (idx >= NR * HH) return;
    const int r = idx >> 9;      // /512
    const int c = idx & 511;
    float s = 0.0f;
    if (r < BB * 7) {
        const int b = r / 7, i = r % 7;
        size_t base = ((size_t)b * 42 + i * 6) * HH + c;
#pragma unroll
        for (int q = 0; q < 6; ++q) s += ef[base + (size_t)q * HH];
    }
    agg[idx] = s;
}

// -------------------- node-MLP input assembly ------------------------------
__global__ void node_input_kernel(const float* __restrict__ flat,
                                  const float* __restrict__ agg,
                                  const int* __restrict__ action,
                                  _Float16* __restrict__ Xn, int round0) {
    const int idx = blockIdx.x * blockDim.x + threadIdx.x;
    if (idx >= NR * KN_PAD) return;
    const int r = idx / KN_PAD, c = idx % KN_PAD;
    float v = 0.0f;
    if (c < DD) {
        v = flat[(size_t)r * DD + c];
    } else if (c < DD + AA) {
        if (round0) { const int b = r >> 3; v = (action[b] == (c - DD)) ? 1.0f : 0.0f; }
    } else if (c < KN_RAW) {
        v = agg[(size_t)r * HH + (c - DD - AA)];
    }
    Xn[idx] = (_Float16)v;
}

// -------------------- residual add / init ----------------------------------
__global__ void copy_kernel(const float* __restrict__ src, float* __restrict__ dst, int n) {
    const int i = blockIdx.x * blockDim.x + threadIdx.x;
    if (i < n) dst[i] = src[i];
}
__global__ void add_kernel(float* __restrict__ flat, const float* __restrict__ node, int n) {
    const int i = blockIdx.x * blockDim.x + threadIdx.x;
    if (i < n) flat[i] += node[i];
}

// ===========================================================================
extern "C" void kernel_launch(void* const* d_in, const int* in_sizes, int n_in,
                              void* d_out, int out_size, void* d_ws, size_t ws_size,
                              hipStream_t stream) {
    const float* states = (const float*)d_in[0];
    const int*   action = (const int*)d_in[1];
    const float* e_w1 = (const float*)d_in[2];  const float* e_b1 = (const float*)d_in[3];
    const float* e_w2 = (const float*)d_in[4];  const float* e_b2 = (const float*)d_in[5];
    const float* e_g  = (const float*)d_in[6];  const float* e_bn = (const float*)d_in[7];
    const float* e_w3 = (const float*)d_in[8];  const float* e_b3 = (const float*)d_in[9];
    const float* n_w1 = (const float*)d_in[10]; const float* n_b1 = (const float*)d_in[11];
    const float* n_w2 = (const float*)d_in[12]; const float* n_b2 = (const float*)d_in[13];
    const float* n_g  = (const float*)d_in[14]; const float* n_bn = (const float*)d_in[15];
    const float* n_w3 = (const float*)d_in[16]; const float* n_b3 = (const float*)d_in[17];

    float* flat = (float*)d_out; // [8192, 256] master state, updated in place

    // ---- workspace carve-out (256B aligned) ----
    char* wp = (char*)d_ws;
    auto alloc = [&](size_t bytes) -> void* {
        void* p = (void*)wp;
        wp += (bytes + 255) & ~(size_t)255;
        return p;
    };
    _Float16* e_w1t = (_Float16*)alloc((size_t)HH * KE * 2);      // [512,768]
    _Float16* e_w2t = (_Float16*)alloc((size_t)HH * HH * 2);      // [512,512]
    _Float16* e_w3t = (_Float16*)alloc((size_t)HH * HH * 2);      // [512,512]
    _Float16* n_w1t = (_Float16*)alloc((size_t)HH * KN_PAD * 2);  // [512,800]
    _Float16* n_w2t = (_Float16*)alloc((size_t)HH * HH * 2);      // [512,512]
    _Float16* n_w3t = (_Float16*)alloc((size_t)DD * HH * 2);      // [256,512]
    _Float16* Xe    = (_Float16*)alloc((size_t)NE * KE * 2);      // edge inputs
    _Float16* h1h   = (_Float16*)alloc((size_t)NE * HH * 2);      // also reused as h2h
    float*    h2f   = (float*)   alloc((size_t)NE * HH * 4);      // also reused as ef
    float*    aggf  = (float*)   alloc((size_t)NR * HH * 4);
    _Float16* Xn    = (_Float16*)alloc((size_t)NR * KN_PAD * 2);
    _Float16* nh1   = (_Float16*)alloc((size_t)NR * HH * 2);      // also reused as nh2h
    float*    nh2f  = (float*)   alloc((size_t)NR * HH * 4);      // also reused as node
    _Float16* h2h   = h1h;
    float*    ef    = h2f;
    _Float16* nh2h  = nh1;
    float*    nodef = nh2f;     // [8192,256] aliases nh2f (8.4MB <= 16.8MB)
    (void)ws_size; (void)n_in; (void)in_sizes; (void)out_size;

    // ---- init state & weights (f16, transposed, K padded) ----
    copy_kernel<<<(NR * DD + 255) / 256, 256, 0, stream>>>(states, flat, NR * DD);
    wtrans_kernel<<<(HH * KE + 255) / 256, 256, 0, stream>>>(e_w1, e_w1t, KE, HH, KE);
    wtrans_kernel<<<(HH * HH + 255) / 256, 256, 0, stream>>>(e_w2, e_w2t, HH, HH, HH);
    wtrans_kernel<<<(HH * HH + 255) / 256, 256, 0, stream>>>(e_w3, e_w3t, HH, HH, HH);
    wtrans_kernel<<<(HH * KN_PAD + 255) / 256, 256, 0, stream>>>(n_w1, n_w1t, KN_RAW, HH, KN_PAD);
    wtrans_kernel<<<(HH * HH + 255) / 256, 256, 0, stream>>>(n_w2, n_w2t, HH, HH, HH);
    wtrans_kernel<<<(DD * HH + 255) / 256, 256, 0, stream>>>(n_w3, n_w3t, HH, DD, HH);

    const dim3 gemmBlk(256, 1, 1);
    const dim3 gE(HH / 64, NE / 256, 1);   // (8, 168)  edge GEMMs, N=512
    const dim3 gN1(HH / 64, NR / 256, 1);  // (8, 32)   node GEMMs, N=512
    const dim3 gN3(DD / 64, NR / 256, 1);  // (4, 32)   node out GEMM, N=256

    for (int round = 0; round < NN; ++round) {
        const int round0 = (round == 0) ? 1 : 0;

        // ---- edge MLP ----
        edge_input_kernel<<<(int)(((size_t)NE * KE + 255) / 256), 256, 0, stream>>>(
            flat, nodef, Xe, round0);
        gemm_wmma_kernel<<<gE, gemmBlk, 0, stream>>>(
            Xe, e_w1t, e_b1, nullptr, h1h, NE, HH, KE, 1);          // relu -> f16
        gemm_wmma_kernel<<<gE, gemmBlk, 0, stream>>>(
            h1h, e_w2t, e_b2, h2f, nullptr, NE, HH, HH, 0);         // f32 pre-LN
        ln_relu_kernel<<<NE, 256, 0, stream>>>(h2f, e_g, e_bn, h2h);
        gemm_wmma_kernel<<<gE, gemmBlk, 0, stream>>>(
            h2h, e_w3t, e_b3, ef, nullptr, NE, HH, HH, 0);          // f32 edge out

        // ---- aggregate (contiguous segment sum) ----
        agg_kernel<<<(NR * HH + 255) / 256, 256, 0, stream>>>(ef, aggf);

        // ---- node MLP ----
        node_input_kernel<<<(NR * KN_PAD + 255) / 256, 256, 0, stream>>>(
            flat, aggf, action, Xn, round0);
        gemm_wmma_kernel<<<gN1, gemmBlk, 0, stream>>>(
            Xn, n_w1t, n_b1, nullptr, nh1, NR, HH, KN_PAD, 1);      // relu -> f16
        gemm_wmma_kernel<<<gN1, gemmBlk, 0, stream>>>(
            nh1, n_w2t, n_b2, nh2f, nullptr, NR, HH, HH, 0);        // f32 pre-LN
        ln_relu_kernel<<<NR, 256, 0, stream>>>(nh2f, n_g, n_bn, nh2h);
        gemm_wmma_kernel<<<gN3, gemmBlk, 0, stream>>>(
            nh2h, n_w3t, n_b3, nodef, nullptr, NR, DD, HH, 0);      // f32 node out

        // ---- residual ----
        add_kernel<<<(NR * DD + 255) / 256, 256, 0, stream>>>(flat, nodef, NR * DD);
    }
}
